// GATNet_8993661518250
// MI455X (gfx1250) — matrix operand order repaired and checked by reading, not compiled, not run on hardware
//
#include <hip/hip_runtime.h>
#include <math.h>

// ---------------- problem constants ----------------
#define N_NODES 20000
#define N_EDGES 320000
#define E_TOT   (N_EDGES + N_NODES)   // + self loops
#define F_IN    128
#define HID     256
#define HEADS   4
#define D_HEAD  64
#define N_GRAPHS 128
#define N_CLS   10
#define EPS     1e-5f
#define NEG_SLOPE 0.2f

typedef _Float16 f16;
typedef __attribute__((ext_vector_type(8)))  _Float16 v8h;
typedef __attribute__((ext_vector_type(16))) _Float16 v16h;
typedef __attribute__((ext_vector_type(8)))  float    v8f;

// ---------------- helpers ----------------
__device__ inline void atomicMaxF(float* addr, float val) {
  if (val >= 0.0f) atomicMax((int*)addr, __float_as_int(val));
  else             atomicMin((unsigned int*)addr, __float_as_uint(val));
}

// ---------------- generic small kernels ----------------
__global__ void k_fill(float* p, float v, int n) {
  int i = blockIdx.x * blockDim.x + threadIdx.x;
  if (i < n) p[i] = v;
}

__global__ void k_cvt_f16(const float* __restrict__ s, f16* __restrict__ d, int n) {
  int i = blockIdx.x * blockDim.x + threadIdx.x;
  if (i < n) d[i] = (f16)s[i];
}

// column mean / biased var of X [M][N]; one block per column
__global__ void k_colstats(const float* __restrict__ X, int M, int N,
                           float* __restrict__ mean, float* __restrict__ var) {
  int col = blockIdx.x, t = threadIdx.x;
  __shared__ float s1[256], s2[256];
  float a = 0.f, b = 0.f;
  for (int r = t; r < M; r += 256) {
    float v = X[(size_t)r * N + col];
    a += v; b += v * v;
  }
  s1[t] = a; s2[t] = b; __syncthreads();
  for (int off = 128; off > 0; off >>= 1) {
    if (t < off) { s1[t] += s1[t + off]; s2[t] += s2[t + off]; }
    __syncthreads();
  }
  if (t == 0) {
    float m = s1[0] / (float)M;
    mean[col] = m;
    var[col]  = s2[0] / (float)M - m * m;
  }
}

// Fold BN into GEMM: Wt[n][k] = s_k*W[k][n] (f16 and/or f32), cp[n] = sum_k t_k*W[k][n] (+bias)
__global__ void k_fold(const float* __restrict__ W, const float* __restrict__ mean,
                       const float* __restrict__ var, const float* __restrict__ gamma,
                       const float* __restrict__ beta, const float* __restrict__ extra_bias,
                       int K, int N, f16* WtH, float* WtF, float* cp) {
  int n = blockIdx.x, t = threadIdx.x;
  __shared__ float red[256];
  float partial = 0.f;
  for (int k = t; k < K; k += 256) {
    float s  = gamma[k] * rsqrtf(var[k] + EPS);
    float tt = beta[k] - mean[k] * s;
    float w  = W[(size_t)k * N + n];
    float ws = s * w;
    if (WtH) WtH[(size_t)n * K + k] = (f16)ws;
    if (WtF) WtF[(size_t)n * K + k] = ws;
    partial += tt * w;
  }
  red[t] = partial; __syncthreads();
  for (int off = 128; off > 0; off >>= 1) {
    if (t < off) red[t] += red[t + off];
    __syncthreads();
  }
  if (t == 0) cp[n] = red[0] + (extra_bias ? extra_bias[n] : 0.f);
}

// ---------------- WMMA GEMM ----------------
// out[M][N] = A[M][K](f16) * Bt[N][K]^T + cp ; optional relu and f16 copy.
// Block: 256 threads = 8 waves. Block tile: 128 rows (8 M-tiles) x 64 cols.
// B panel (64 x K, f16) staged in LDS once per block; each wave register-blocks
// 4 N-tiles on one M-tile, so each A fragment feeds 4 WMMAs.
#define BN_COLS 64
__global__ void k_gemm_wmma(const f16* __restrict__ A, const f16* __restrict__ Bt,
                            const float* __restrict__ cp, float* __restrict__ outF,
                            f16* __restrict__ outH, int M, int N, int K, int relu) {
  __shared__ f16 Bs[BN_COLS * 256];        // supports K <= 256 (32 KB)
  int t = threadIdx.x;
  int colbase = blockIdx.x * BN_COLS;

  // cooperative B panel load: 64*K halfs in 16B chunks
  int nchunks = (BN_COLS * K) >> 3;        // /8 halfs per chunk
  for (int c = t; c < nchunks; c += 256) {
    int off = c << 3;
    int lc = off / K, k = off % K;
    *(v8h*)(&Bs[lc * K + k]) = *(const v8h*)(Bt + (size_t)(colbase + lc) * K + k);
  }
  __syncthreads();

  int wave = t >> 5;
  int lane = t & 31;
  int mtiles = M >> 4;
  int mt = blockIdx.y * 8 + wave;          // wave-uniform
  if (mt >= mtiles) return;                // whole-wave exit; EXEC all-ones below

  int hi = lane >> 4;
  int mrow = (mt << 4) + (lane & 15);
  const f16* arow = A + (size_t)mrow * K;

  v8f acc0 = {}, acc1 = {}, acc2 = {}, acc3 = {};
  for (int kk = 0; kk < K; kk += 32) {
    int ka = kk + (hi << 4);               // A: 16 contiguous K per lane
    if (kk + 32 < K) __builtin_prefetch(arow + ka + 32, 0, 0);  // global_prefetch_b8
    v8h alo = *(const v8h*)(arow + ka);
    v8h ahi = *(const v8h*)(arow + ka + 8);
    v16h a = __builtin_shufflevector(alo, ahi,
             0,1,2,3,4,5,6,7,8,9,10,11,12,13,14,15);
    int kb = kk + (hi << 4);               // B: 16 contiguous K per lane
    int lc = lane & 15;
    v16h b0 = *(const v16h*)(&Bs[(lc      ) * K + kb]);
    v16h b1 = *(const v16h*)(&Bs[(lc + 16 ) * K + kb]);
    v16h b2 = *(const v16h*)(&Bs[(lc + 32 ) * K + kb]);
    v16h b3 = *(const v16h*)(&Bs[(lc + 48 ) * K + kb]);
    acc0 = __builtin_amdgcn_wmma_f32_16x16x32_f16(false, a, false, b0, (short)0, acc0, false, false);
    acc1 = __builtin_amdgcn_wmma_f32_16x16x32_f16(false, a, false, b1, (short)0, acc1, false, false);
    acc2 = __builtin_amdgcn_wmma_f32_16x16x32_f16(false, a, false, b2, (short)0, acc2, false, false);
    acc3 = __builtin_amdgcn_wmma_f32_16x16x32_f16(false, a, false, b3, (short)0, acc3, false, false);
  }

  v8f accs[4] = {acc0, acc1, acc2, acc3};
  #pragma unroll
  for (int tt = 0; tt < 4; ++tt) {
    int col = colbase + tt * 16 + (lane & 15);
    float cb = cp[col];
    #pragma unroll
    for (int j = 0; j < 8; ++j) {          // D layout: M=j+8*hi, N=lane&15
      int r = (mt << 4) + j + (hi << 3);
      float v = accs[tt][j] + cb;
      if (relu) v = fmaxf(v, 0.f);
      size_t o = (size_t)r * N + col;
      outF[o] = v;
      if (outH) outH[o] = (f16)v;
    }
  }
}

// ---------------- attention ----------------
__global__ void k_attn(const float* __restrict__ xp, const float* __restrict__ asrc,
                       const float* __restrict__ adst, float* __restrict__ als,
                       float* __restrict__ ald) {
  int idx = blockIdx.x * blockDim.x + threadIdx.x;
  if (idx >= N_NODES * HEADS) return;
  int n = idx >> 2, h = idx & 3;
  const float* xr = xp + (size_t)n * HID + h * D_HEAD;
  const float* as = asrc + h * D_HEAD;
  const float* ad = adst + h * D_HEAD;
  float s0 = 0.f, s1 = 0.f;
  #pragma unroll 8
  for (int d = 0; d < D_HEAD; ++d) { float v = xr[d]; s0 += v * as[d]; s1 += v * ad[d]; }
  als[idx] = s0; ald[idx] = s1;
}

__global__ void k_edge_max(const int* __restrict__ ei, const float* __restrict__ als,
                           const float* __restrict__ ald, float* __restrict__ mx,
                           float* __restrict__ elog) {
  int idx = blockIdx.x * blockDim.x + threadIdx.x;
  if (idx >= E_TOT * HEADS) return;
  int e = idx >> 2, h = idx & 3;
  int s, d;
  if (e < N_EDGES) { s = ei[e]; d = ei[N_EDGES + e]; } else { s = d = e - N_EDGES; }
  float l = als[s * HEADS + h] + ald[d * HEADS + h];
  float lr = l > 0.f ? l : NEG_SLOPE * l;
  elog[idx] = lr;
  atomicMaxF(&mx[d * HEADS + h], lr);
}

__global__ void k_edge_exp(const int* __restrict__ ei, const float* __restrict__ mx,
                           float* __restrict__ elog, float* __restrict__ ssum) {
  int idx = blockIdx.x * blockDim.x + threadIdx.x;
  if (idx >= E_TOT * HEADS) return;
  int e = idx >> 2, h = idx & 3;
  int d = (e < N_EDGES) ? ei[N_EDGES + e] : (e - N_EDGES);
  float p = expf(elog[idx] - mx[d * HEADS + h]);
  elog[idx] = p;
  atomicAdd(&ssum[d * HEADS + h], p);
}

// one wave per edge; dim = lane + 32k => head = k>>1; coalesced 128B per k-step
__global__ void k_edge_scatter(const int* __restrict__ ei, const float* __restrict__ xp,
                               const float* __restrict__ p, const float* __restrict__ ssum,
                               float* __restrict__ acc) {
  int e    = (blockIdx.x * blockDim.x + threadIdx.x) >> 5;
  int lane = threadIdx.x & 31;
  if (e >= E_TOT) return;
  int s, d;
  if (e < N_EDGES) { s = ei[e]; d = ei[N_EDGES + e]; } else { s = d = e - N_EDGES; }
  float al[HEADS];
  #pragma unroll
  for (int h = 0; h < HEADS; ++h)
    al[h] = p[e * HEADS + h] / ssum[d * HEADS + h];
  const float* xs = xp  + (size_t)s * HID;
  float*       ad = acc + (size_t)d * HID;
  #pragma unroll
  for (int k = 0; k < 8; ++k) {
    int dim = lane + 32 * k;
    atomicAdd(&ad[dim], xs[dim] * al[k >> 1]);
  }
}

__global__ void k_bias_relu(float* __restrict__ acc, f16* __restrict__ hlp,
                            const float* __restrict__ bias, int total) {
  int idx = blockIdx.x * blockDim.x + threadIdx.x;
  if (idx >= total) return;
  float v = acc[idx] + bias[idx & (HID - 1)];
  v = fmaxf(v, 0.f);
  acc[idx] = v;
  hlp[idx] = (f16)v;
}

__global__ void k_pool(const float* __restrict__ h, const int* __restrict__ batch,
                       float* __restrict__ g) {
  int idx = blockIdx.x * blockDim.x + threadIdx.x;
  if (idx >= N_NODES * HID) return;
  int n = idx >> 8;
  atomicAdd(&g[(size_t)batch[n] * HID + (idx & (HID - 1))], h[idx]);
}

// final classifier with folded BN + log_softmax; one block (32 thr) per graph
__global__ void k_cls(const float* __restrict__ gfc, const float* __restrict__ WtF,
                      const float* __restrict__ cp, float* __restrict__ out) {
  int g = blockIdx.x, c = threadIdx.x;
  __shared__ float logits[16];
  __shared__ float mx_s, lse_s;
  if (c < N_CLS) {
    float a = cp[c];
    const float* gr = gfc + (size_t)g * HID;
    const float* wr = WtF + (size_t)c * HID;
    for (int k = 0; k < HID; ++k) a += gr[k] * wr[k];
    logits[c] = a;
  }
  __syncthreads();
  if (c == 0) {
    float mx = -1e30f;
    for (int i = 0; i < N_CLS; ++i) mx = fmaxf(mx, logits[i]);
    float se = 0.f;
    for (int i = 0; i < N_CLS; ++i) se += expf(logits[i] - mx);
    mx_s = mx; lse_s = logf(se);
  }
  __syncthreads();
  if (c < N_CLS) out[g * N_CLS + c] = logits[c] - mx_s - lse_s;
}

// ---------------- launcher ----------------
extern "C" void kernel_launch(void* const* d_in, const int* in_sizes, int n_in,
                              void* d_out, int out_size, void* d_ws, size_t ws_size,
                              hipStream_t stream) {
  const float* x        = (const float*)d_in[0];
  const int*   ei       = (const int*)d_in[1];
  const int*   batch    = (const int*)d_in[2];
  const float* bnf_g    = (const float*)d_in[3];
  const float* bnf_b    = (const float*)d_in[4];
  const float* W_feat   = (const float*)d_in[5];
  const float* b_feat   = (const float*)d_in[6];
  const float* bns_g    = (const float*)d_in[7];
  const float* bns_b    = (const float*)d_in[8];
  const float* Wg       = (const float*)d_in[9];
  const float* att_src  = (const float*)d_in[10];
  const float* att_dst  = (const float*)d_in[11];
  const float* gat_b    = (const float*)d_in[12];
  const float* bnfc_g   = (const float*)d_in[13];
  const float* bnfc_b   = (const float*)d_in[14];
  const float* W_fc     = (const float*)d_in[15];
  const float* b_fc     = (const float*)d_in[16];
  const float* bnh_g    = (const float*)d_in[17];
  const float* bnh_b    = (const float*)d_in[18];
  const float* W_cls    = (const float*)d_in[19];
  const float* b_cls    = (const float*)d_in[20];
  float* out = (float*)d_out;

  // workspace carve-up (256B aligned)
  char* p = (char*)d_ws;
  auto alloc = [&](size_t bytes) -> void* {
    void* r = (void*)p; p += (bytes + 255) & ~(size_t)255; return r;
  };
  f16*   x_lp  = (f16*)  alloc((size_t)N_NODES * F_IN * 2);
  f16*   h_lp  = (f16*)  alloc((size_t)N_NODES * HID * 2);
  float* hA    = (float*)alloc((size_t)N_NODES * HID * 4);   // h f32, reused as scatter acc
  float* xp    = (float*)alloc((size_t)N_NODES * HID * 4);
  float* als   = (float*)alloc((size_t)N_NODES * HEADS * 4);
  float* ald   = (float*)alloc((size_t)N_NODES * HEADS * 4);
  float* mx    = (float*)alloc((size_t)N_NODES * HEADS * 4);
  float* ssum  = (float*)alloc((size_t)N_NODES * HEADS * 4);
  float* elog  = (float*)alloc((size_t)E_TOT * HEADS * 4);
  float* meanb = (float*)alloc(HID * 4);
  float* varb  = (float*)alloc(HID * 4);
  f16*   WtH   = (f16*)  alloc((size_t)HID * HID * 2);
  float* WtF   = (float*)alloc((size_t)N_CLS * HID * 4);
  float* cpb   = (float*)alloc(HID * 4);
  float* gpool = (float*)alloc((size_t)N_GRAPHS * HID * 4);
  f16*   g_lp  = (f16*)  alloc((size_t)N_GRAPHS * HID * 2);
  float* gfc   = (float*)alloc((size_t)N_GRAPHS * HID * 4);

  auto cdiv = [](int a, int b) { return (a + b - 1) / b; };
  const int NH   = N_NODES * HEADS;       // 80000
  const int EH   = E_TOT * HEADS;         // 1360000
  const int NHID = N_NODES * HID;         // 5120000
  const int MT_N = N_NODES / 16;          // 1250 M-tiles
  const int MT_G = N_GRAPHS / 16;         // 8 M-tiles

  // ---- stage 0: BN(x) folded into W_feat, h = relu(x @ W' + c') ----
  k_colstats<<<F_IN, 256, 0, stream>>>(x, N_NODES, F_IN, meanb, varb);
  k_fold<<<HID, 256, 0, stream>>>(W_feat, meanb, varb, bnf_g, bnf_b, b_feat,
                                  F_IN, HID, WtH, nullptr, cpb);
  k_cvt_f16<<<cdiv(N_NODES * F_IN, 256), 256, 0, stream>>>(x, x_lp, N_NODES * F_IN);
  k_gemm_wmma<<<dim3(HID / BN_COLS, cdiv(MT_N, 8)), 256, 0, stream>>>(
      x_lp, WtH, cpb, hA, h_lp, N_NODES, HID, F_IN, 1);

  // ---- 3 GAT layers ----
  for (int i = 0; i < 3; ++i) {
    k_colstats<<<HID, 256, 0, stream>>>(hA, N_NODES, HID, meanb, varb);
    k_fold<<<HID, 256, 0, stream>>>(Wg + (size_t)i * HID * HID, meanb, varb,
                                    bns_g + i * HID, bns_b + i * HID, nullptr,
                                    HID, HID, WtH, nullptr, cpb);
    k_gemm_wmma<<<dim3(HID / BN_COLS, cdiv(MT_N, 8)), 256, 0, stream>>>(
        h_lp, WtH, cpb, xp, nullptr, N_NODES, HID, HID, 0);
    k_attn<<<cdiv(NH, 256), 256, 0, stream>>>(xp, att_src + i * HID, att_dst + i * HID,
                                              als, ald);
    k_fill<<<cdiv(NH, 256), 256, 0, stream>>>(mx, -INFINITY, NH);
    k_fill<<<cdiv(NH, 256), 256, 0, stream>>>(ssum, 0.f, NH);
    k_fill<<<cdiv(NHID, 256), 256, 0, stream>>>(hA, 0.f, NHID);   // hA becomes scatter acc
    k_edge_max<<<cdiv(EH, 256), 256, 0, stream>>>(ei, als, ald, mx, elog);
    k_edge_exp<<<cdiv(EH, 256), 256, 0, stream>>>(ei, mx, elog, ssum);
    k_edge_scatter<<<cdiv(E_TOT * 32, 256), 256, 0, stream>>>(ei, xp, elog, ssum, hA);
    k_bias_relu<<<cdiv(NHID, 256), 256, 0, stream>>>(hA, h_lp, gat_b + i * HID, NHID);
  }

  // ---- pool + head ----
  k_fill<<<cdiv(N_GRAPHS * HID, 256), 256, 0, stream>>>(gpool, 0.f, N_GRAPHS * HID);
  k_pool<<<cdiv(NHID, 256), 256, 0, stream>>>(hA, batch, gpool);

  k_colstats<<<HID, 256, 0, stream>>>(gpool, N_GRAPHS, HID, meanb, varb);
  k_fold<<<HID, 256, 0, stream>>>(W_fc, meanb, varb, bnfc_g, bnfc_b, b_fc,
                                  HID, HID, WtH, nullptr, cpb);
  k_cvt_f16<<<cdiv(N_GRAPHS * HID, 256), 256, 0, stream>>>(gpool, g_lp, N_GRAPHS * HID);
  k_gemm_wmma<<<dim3(HID / BN_COLS, cdiv(MT_G, 8)), 256, 0, stream>>>(
      g_lp, WtH, cpb, gfc, nullptr, N_GRAPHS, HID, HID, 1);

  k_colstats<<<HID, 256, 0, stream>>>(gfc, N_GRAPHS, HID, meanb, varb);
  k_fold<<<N_CLS, 256, 0, stream>>>(W_cls, meanb, varb, bnh_g, bnh_b, b_cls,
                                    HID, N_CLS, nullptr, WtF, cpb);
  k_cls<<<N_GRAPHS, 32, 0, stream>>>(gfc, WtF, cpb, out);
  (void)in_sizes; (void)n_in; (void)out_size; (void)ws_size;
}
